// MAESelfAttention_31447750541649
// MI455X (gfx1250) — compile-verified, matrix operand order checked by reading, not run on hardware
//
#include <hip/hip_runtime.h>
#include <hip/hip_bf16.h>
#include <cstddef>
#include <cstdint>

typedef __attribute__((ext_vector_type(16))) _Float16     v16h;
typedef __attribute__((ext_vector_type(8)))  float        v8f;
typedef __attribute__((ext_vector_type(4)))  unsigned int v4u;
typedef __attribute__((ext_vector_type(8)))  int          v8i;
typedef __attribute__((ext_vector_type(4)))  int          v4i;

#define BB    2
#define SS    2048
#define EE    8
#define HID_  768
#define NH_   12
#define HD_   64
#define LTOT  2056   // E + S
#define NEGB  (-10000.0f)

// TDM availability (this toolchain exposes the 6-arg clang-23 form)
#if defined(__has_builtin)
#if __has_builtin(__builtin_amdgcn_tensor_load_to_lds) && \
    __has_builtin(__builtin_amdgcn_s_wait_tensorcnt)
#define USE_TDM 1
#endif
#endif
#ifndef USE_TDM
#define USE_TDM 0
#endif

// Fragment viewed either as 16 halves or 8 packed dwords (VGPR granularity)
union F16Frag { v16h h; uint32_t u[8]; };

// K index base for 16-bit A/B fragments: VGPR v holds pair {k, k+1}
__device__ __forceinline__ int kmap(int v, int half) {
    int base = (v < 4) ? (2 * v) : (16 + 2 * (v - 4));
    return base + 8 * half;
}

// emb[p, c] of the reference's _sinusoidal(MAXP, 64)
__device__ __forceinline__ float embv(int p, int c) {
    int j = c >> 1;
    float ang = (float)p * __expf(-0.14391156831212787f * (float)(2 * j)); // ln(1e4)/64
    return (c & 1) ? __cosf(ang) : __sinf(ang);
}

// -------------------- Kernel 1: RoPE(Q) -> f16 (B,NH,S,64) --------------------
__global__ __launch_bounds__(256) void qprep_kernel(const float* __restrict__ qh,
                                                    _Float16* __restrict__ qbuf) {
    size_t idx = (size_t)blockIdx.x * 256 + threadIdx.x;   // total B*NH*S*64
    int d = (int)(idx & 63);
    int s = (int)((idx >> 6) & 2047);
    int bh = (int)(idx >> 17);          // b*12 + h
    int h = bh % NH_;
    int b = bh / NH_;
    const float* row = qh + ((size_t)b * SS + s) * HID_ + h * HD_;
    float partner = row[d ^ 1];
    float r = (d & 1) ? partner * embv(s, (d - 1) >> 1)
                      : -partner * embv(s, 32 + (d >> 1));
    qbuf[idx] = (_Float16)r;
}

// ---------- Kernel 2: KV projection GEMM (WMMA) + RoPE(K) epilogue ----------
// grid (257, 96), block 32 (1 wave -> one 16x16 output tile of kv = ekv @ W^T)
__global__ __launch_bounds__(32) void kvproj_kernel(const float* __restrict__ embx,
                                                    const float* __restrict__ kvh,
                                                    const float* __restrict__ W,
                                                    const float* __restrict__ bvec,
                                                    _Float16* __restrict__ kbuf,
                                                    _Float16* __restrict__ vbuf) {
    const int lane = threadIdx.x & 31;
    const int half = lane >> 4;
    const int ln   = lane & 15;
    const int mt = blockIdx.x;   // 257 tiles of 16 rows (B*(E+S) = 4112)
    const int nt = blockIdx.y;   // 96 tiles of 16 cols (2*HID = 1536)

    int mrow = mt * 16 + ln;
    int bA   = mrow / LTOT;
    int pos  = mrow % LTOT;
    const float* arow = (pos < EE)
        ? (embx + ((size_t)bA * EE + pos) * HID_)
        : (kvh  + ((size_t)bA * SS + (pos - EE)) * HID_);
    int ncol = nt * 16 + ln;
    const float* wrow = W + (size_t)ncol * HID_;

    v8f acc = {};
    for (int k0 = 0; k0 < HID_; k0 += 32) {
        v16h afrag, wfrag;
#pragma unroll
        for (int v = 0; v < 8; ++v) {
            int kk = k0 + kmap(v, half);
            float2 fa = *(const float2*)(arow + kk);
            float2 fw = *(const float2*)(wrow + kk);
            afrag[2 * v]     = (_Float16)fa.x;
            afrag[2 * v + 1] = (_Float16)fa.y;
            wfrag[2 * v]     = (_Float16)fw.x;
            wfrag[2 * v + 1] = (_Float16)fw.y;
        }
        acc = __builtin_amdgcn_wmma_f32_16x16x32_f16(false, afrag, false, wfrag,
                                                     (short)0, acc, false, false);
    }

    // Epilogue: C layout M = v + 8*half, N = ln. Columns within a tile never
    // straddle the K/V or head boundary (16 | 64 | 768).
#pragma unroll
    for (int v = 0; v < 8; ++v) {
        int mr = mt * 16 + v + 8 * half;
        int bO = mr / LTOT;
        int pp = mr % LTOT;
        int nc = nt * 16 + ln;
        float val = acc[v] + bvec[nc];
        // lanes ln and ln^1 hold d and d^1 of the same row -> exchange for RoPE
        float partner = __shfl_xor(val, 1, 32);
        int c = nc / HID_;
        int h = (nc >> 6) % NH_;
        int d = nc & 63;
        if (c == 0) {
            float r = (d & 1) ? partner * embv(pp, (d - 1) >> 1)
                              : -partner * embv(pp, 32 + (d >> 1));
            kbuf[(((size_t)bO * NH_ + h) * LTOT + pp) * HD_ + d] = (_Float16)r;
        } else {
            vbuf[(((size_t)bO * NH_ + h) * LTOT + pp) * HD_ + d] = (_Float16)val;
        }
    }
}

// ----------------- Kernel 3: flash attention with inline bias -----------------
__device__ __forceinline__ float biasval(int b, int q, int key,
                                         const float* __restrict__ randu,
                                         const unsigned char* __restrict__ amask) {
    if (key >= LTOT) return -1.0e30f;           // tail padding
    if (key < EE) return 0.0f;                  // prepended emb tokens
    int kk = key - EE;
    bool enh = (amask[(size_t)b * SS + kk] != 0) &&
               (randu[((size_t)b * SS + q) * SS + kk] >= 0.1f) &&
               (q != kk);
    return enh ? 0.0f : NEGB;
}

#if USE_TDM
// TDM: load a (rows_remaining clamped to 32) x 64 f16 tile, row stride 64, into LDS.
// 2-D descriptor: groups 0/1 populated, groups 2/3 (and trailing group arg) zero.
__device__ __forceinline__ void tdm_load_tile(uint32_t lds_addr,
                                              const _Float16* gsrc,
                                              int rows_remaining) {
    uint64_t ga = (uint64_t)(uintptr_t)gsrc;
    v4u g0;
    g0[0] = 1u;                                   // count=1, user mode
    g0[1] = lds_addr;                             // lds_addr [63:32]
    g0[2] = (uint32_t)ga;                         // global_addr lo [95:64]
    g0[3] = (uint32_t)(ga >> 32) | 0x80000000u;   // global_addr hi | type=2 [127:126]
    v8i g1;
    g1[0] = (1 << 16);                            // data_size=1 -> 2 bytes
    g1[1] = (int)(64u << 16);                     // tensor_dim0=64 (bits 79:48)
    g1[2] = (int)(((uint32_t)rows_remaining & 0xFFFFu) << 16);  // tensor_dim1 lo (111:80)
    g1[3] = (int)((((uint32_t)rows_remaining >> 16) & 0xFFFFu)
                  | (64u << 16));                 // tensor_dim1 hi | tile_dim0=64 (127:112)
    g1[4] = 32;                                   // tile_dim1=32 (143:128), tile_dim2=0
    g1[5] = 64;                                   // tensor_dim0_stride lo (191:160)
    g1[6] = 0;                                    // stride0 hi / stride1 lo
    g1[7] = 0;                                    // stride1 hi
    v4i z4 = {0, 0, 0, 0};
    v8i z8 = {0, 0, 0, 0, 0, 0, 0, 0};
    __builtin_amdgcn_tensor_load_to_lds(g0, g1, z4, z4, z8, 0);
}
#endif

// grid (S/64, NH, B), block 128 (4 waves x 16 queries)
__global__ __launch_bounds__(128) void flash_kernel(const _Float16* __restrict__ qbuf,
                                                    const _Float16* __restrict__ kbuf,
                                                    const _Float16* __restrict__ vbuf,
                                                    const float* __restrict__ randu,
                                                    const unsigned char* __restrict__ amask,
                                                    float* __restrict__ out) {
    __shared__ _Float16 Kt[32 * 64];
    __shared__ _Float16 Vt[32 * 64];
    __shared__ _Float16 Pst[4][16 * 32];

    const int tid  = threadIdx.x;
    const int wid  = tid >> 5;
    const int lane = tid & 31;
    const int half = lane >> 4;
    const int ln   = lane & 15;
    const int h = blockIdx.y;
    const int b = blockIdx.z;
    const int q0 = blockIdx.x * 64 + wid * 16;

    const _Float16* qbase = qbuf + (((size_t)b * NH_ + h) * SS) * HD_;
    const _Float16* kbase = kbuf + (((size_t)b * NH_ + h) * LTOT) * HD_;
    const _Float16* vbase = vbuf + (((size_t)b * NH_ + h) * LTOT) * HD_;

    // Q A-fragments: d chunks [0,32) and [32,64); f16 pairs are contiguous
    F16Frag qa0, qa1;
    {
        const _Float16* qrow = qbase + (size_t)(q0 + ln) * HD_;
#pragma unroll
        for (int v = 0; v < 8; ++v) {
            int kk = kmap(v, half);
            qa0.u[v] = *(const uint32_t*)(qrow + kk);
            qa1.u[v] = *(const uint32_t*)(qrow + 32 + kk);
        }
    }

    v8f o[4];
#pragma unroll
    for (int c = 0; c < 4; ++c) o[c] = v8f{};
    float mrun[8], lrun[8];
#pragma unroll
    for (int v = 0; v < 8; ++v) { mrun[v] = -3.0e38f; lrun[v] = 0.0f; }

#if USE_TDM
    const uint32_t KtAddr = (uint32_t)(uintptr_t)&Kt[0];
    const uint32_t VtAddr = (uint32_t)(uintptr_t)&Vt[0];
#endif

    for (int kb = 0; kb < LTOT; kb += 32) {
#if USE_TDM
        // Tensor Data Mover: one DMA per tile, issued by wave 0; tail rows are
        // zero-filled by TDM OOB handling (tensor_dim1 = rows remaining).
        if (wid == 0) {
            tdm_load_tile(KtAddr, kbase + (size_t)kb * HD_, LTOT - kb);
            tdm_load_tile(VtAddr, vbase + (size_t)kb * HD_, LTOT - kb);
            __builtin_amdgcn_s_wait_tensorcnt(0);
        }
#else
        // Fallback: cooperative vector load, tail-clamped (bias masks dupes).
#pragma unroll
        for (int i = 0; i < 8; ++i) {
            int idx = tid + i * 128;
            int r  = idx >> 5;
            int cg = idx & 31;
            int src = kb + r; if (src > LTOT - 1) src = LTOT - 1;
            ((uint32_t*)Kt)[r * 32 + cg] = *(const uint32_t*)(kbase + (size_t)src * HD_ + cg * 2);
            ((uint32_t*)Vt)[r * 32 + cg] = *(const uint32_t*)(vbase + (size_t)src * HD_ + cg * 2);
        }
#endif
        __syncthreads();

        // S = Q K^T for two 16-key groups (B-frag d pairs contiguous -> b32 LDS loads)
        v8f s0 = {}, s1 = {};
        {
            F16Frag b0, b1, b2, b3;
#pragma unroll
            for (int v = 0; v < 8; ++v) {
                int kk = kmap(v, half);     // d index within 32-chunk
                b0.u[v] = *(const uint32_t*)(Kt + ln * 64 + kk);
                b1.u[v] = *(const uint32_t*)(Kt + ln * 64 + 32 + kk);
                b2.u[v] = *(const uint32_t*)(Kt + (16 + ln) * 64 + kk);
                b3.u[v] = *(const uint32_t*)(Kt + (16 + ln) * 64 + 32 + kk);
            }
            s0 = __builtin_amdgcn_wmma_f32_16x16x32_f16(false, qa0.h, false, b0.h, (short)0, s0, false, false);
            s0 = __builtin_amdgcn_wmma_f32_16x16x32_f16(false, qa1.h, false, b1.h, (short)0, s0, false, false);
            s1 = __builtin_amdgcn_wmma_f32_16x16x32_f16(false, qa0.h, false, b2.h, (short)0, s1, false, false);
            s1 = __builtin_amdgcn_wmma_f32_16x16x32_f16(false, qa1.h, false, b3.h, (short)0, s1, false, false);
        }

        // bias + online softmax; C rows: M = v + 8*half, cols: N = ln
        const float scl = 0.125f;   // 1/sqrt(64)
#pragma unroll
        for (int v = 0; v < 8; ++v) {
            int qrow = q0 + v + 8 * half;
            s0[v] = s0[v] * scl + biasval(b, qrow, kb + ln,      randu, amask);
            s1[v] = s1[v] * scl + biasval(b, qrow, kb + 16 + ln, randu, amask);

            float t = fmaxf(s0[v], s1[v]);
            t = fmaxf(t, __shfl_xor(t, 1, 32));
            t = fmaxf(t, __shfl_xor(t, 2, 32));
            t = fmaxf(t, __shfl_xor(t, 4, 32));
            t = fmaxf(t, __shfl_xor(t, 8, 32));
            float mnew = fmaxf(mrun[v], t);
            float sc = __expf(mrun[v] - mnew);
            mrun[v] = mnew;

            float e0 = __expf(s0[v] - mnew);
            float e1 = __expf(s1[v] - mnew);
            float rs = e0 + e1;
            rs += __shfl_xor(rs, 1, 32);
            rs += __shfl_xor(rs, 2, 32);
            rs += __shfl_xor(rs, 4, 32);
            rs += __shfl_xor(rs, 8, 32);
            lrun[v] = lrun[v] * sc + rs;
#pragma unroll
            for (int c = 0; c < 4; ++c) o[c][v] *= sc;

            Pst[wid][(v + 8 * half) * 32 + ln]      = (_Float16)e0;
            Pst[wid][(v + 8 * half) * 32 + 16 + ln] = (_Float16)e1;
        }
        // wave-private LDS relayout of P: wait on DScnt before re-reading
        asm volatile("s_wait_dscnt 0" ::: "memory");

        F16Frag pa;
#pragma unroll
        for (int v = 0; v < 8; ++v) {
            int kk = kmap(v, half);         // key index within 32-chunk
            pa.u[v] = *(const uint32_t*)(&Pst[wid][ln * 32 + kk]);
        }

        // O += P V  over 4 d-chunks of 16 (V rows not contiguous -> scalar gathers)
#pragma unroll
        for (int c = 0; c < 4; ++c) {
            F16Frag vfrag;
#pragma unroll
            for (int v = 0; v < 8; ++v) {
                int kk = kmap(v, half);     // key index
                vfrag.h[2 * v]     = Vt[kk * 64 + c * 16 + ln];
                vfrag.h[2 * v + 1] = Vt[(kk + 1) * 64 + c * 16 + ln];
            }
            o[c] = __builtin_amdgcn_wmma_f32_16x16x32_f16(false, pa.h, false, vfrag.h,
                                                          (short)0, o[c], false, false);
        }
        __syncthreads();   // protect Kt/Vt before next tile load
    }

    // Epilogue: out[b, q, h*64 + d] = O / l
#pragma unroll
    for (int c = 0; c < 4; ++c) {
#pragma unroll
        for (int v = 0; v < 8; ++v) {
            int qrow = q0 + v + 8 * half;
            int d = c * 16 + ln;
            out[((size_t)b * SS + qrow) * HID_ + h * HD_ + d] = o[c][v] / lrun[v];
        }
    }
}

// ------------------------------- launcher -------------------------------
extern "C" void kernel_launch(void* const* d_in, const int* in_sizes, int n_in,
                              void* d_out, int out_size, void* d_ws, size_t ws_size,
                              hipStream_t stream) {
    (void)in_sizes; (void)n_in; (void)out_size; (void)ws_size;
    const float* qh    = (const float*)d_in[0];   // (B,S,HID)
    const float* kvh   = (const float*)d_in[1];   // (B,S,HID)
    const float* embx  = (const float*)d_in[2];   // (B,E,HID)
    const float* randu = (const float*)d_in[3];   // (B,S,S)
    // d_in[4] Wq_w, d_in[5] Wq_b: unused by the reference
    const float* Wkv   = (const float*)d_in[6];   // (2*HID, HID)
    const float* Wkvb  = (const float*)d_in[7];   // (2*HID,)
    const unsigned char* amask = (const unsigned char*)d_in[8];  // (B,S) bool

    const size_t QBYTES = (size_t)BB * NH_ * SS   * HD_ * sizeof(_Float16); // 6.29 MB
    const size_t KBYTES = (size_t)BB * NH_ * LTOT * HD_ * sizeof(_Float16); // 6.32 MB
    _Float16* qbuf = (_Float16*)d_ws;
    _Float16* kbuf = (_Float16*)((char*)d_ws + QBYTES);
    _Float16* vbuf = (_Float16*)((char*)d_ws + QBYTES + KBYTES);

    // 1) RoPE(Q) -> f16
    {
        size_t total = (size_t)BB * NH_ * SS * HD_;   // 3,145,728
        qprep_kernel<<<(unsigned)(total / 256), 256, 0, stream>>>(qh, qbuf);
    }
    // 2) KV projection + RoPE(K)
    {
        dim3 grid((BB * LTOT) / 16, (2 * HID_) / 16);  // (257, 96)
        kvproj_kernel<<<grid, 32, 0, stream>>>(embx, kvh, Wkv, Wkvb, kbuf, vbuf);
    }
    // 3) flash attention with inline bias
    {
        dim3 grid(SS / 64, NH_, BB);                   // (32, 12, 2)
        flash_kernel<<<grid, 128, 0, stream>>>(qbuf, kbuf, vbuf, randu, amask,
                                               (float*)d_out);
    }
}
